// PointwiseCorrelation_69458211110976
// MI455X (gfx1250) — matrix operand order repaired and checked by reading, not compiled, not run on hardware
//
#include <hip/hip_runtime.h>

#define B_ 8
#define C_ 256
#define H_ 96
#define W_ 96
#define DMAX 4
#define DD 9           // 2*DMAX+1
#define WT 16          // w-tile (GEMM M)
#define KP 264         // padded K stride in LDS (ushorts): bank-conflict avoidance
#define NCOL 24        // stored B columns per vertical offset (WT + 2*DMAX)
#define NTHREADS 288   // 9 waves of 32

typedef __attribute__((ext_vector_type(16))) __bf16 v16bf;
typedef __attribute__((ext_vector_type(8)))  float  v8f;

union Frag { v16bf v; uint4 q[2]; };

__device__ __forceinline__ unsigned short bf16_rne(float x) {
  unsigned int b = __float_as_uint(x);
  unsigned int r = b + 0x7FFFu + ((b >> 16) & 1u);
  return (unsigned short)(r >> 16);
}

// Dekker-style split: x ~= bf16(hi) + bf16(lo), exact residual before 2nd round.
__device__ __forceinline__ void split2(float x, unsigned short &hi, unsigned short &lo) {
  unsigned short h = bf16_rne(x);
  float fh = __uint_as_float(((unsigned int)h) << 16);
  float res = x - fh;           // exact (fh has <=8 mantissa bits)
  hi = h;
  lo = bf16_rne(res);
}

extern "C" __global__ __launch_bounds__(NTHREADS)
void corr_wmma_kernel(const float* __restrict__ FM0, const float* __restrict__ FM1,
                      float* __restrict__ out) {
  const int wb = blockIdx.x;      // 0..5
  const int h  = blockIdx.y;      // 0..95
  const int b  = blockIdx.z;      // 0..7
  const int w0 = wb * WT;
  const int tid = threadIdx.x;

  extern __shared__ unsigned short smem[];
  unsigned short* sAhi = smem;                       // [WT][KP]
  unsigned short* sAlo = sAhi + WT * KP;             // [WT][KP]
  unsigned short* sBhi = sAlo + WT * KP;             // [DD][NCOL][KP]
  unsigned short* sBlo = sBhi + DD * NCOL * KP;      // [DD][NCOL][KP]

  // ---- Stage A: FM0[b, :, h, w0..w0+15] -> sA[m][c] (hi/lo bf16)
  if (tid < C_) {
    const int c = tid;
    const float* src = FM0 + (((size_t)(b * C_ + c) * H_ + h) * W_ + w0);
#pragma unroll
    for (int g = 0; g < 4; ++g) {
      float4 v = *(const float4*)(src + 4 * g);      // w0 % 16 == 0 -> aligned
      float vv[4] = {v.x, v.y, v.z, v.w};
#pragma unroll
      for (int e = 0; e < 4; ++e) {
        int m = 4 * g + e;
        unsigned short hi, lo;
        split2(vv[e], hi, lo);
        sAhi[m * KP + c] = hi;
        sAlo[m * KP + c] = lo;
      }
    }
  }

  // ---- Stage B band: FM1p[b, :, h+i-4, w0-4 .. w0+19] -> sB[i][n][c]
#pragma unroll
  for (int rep = 0; rep < 8; ++rep) {
    int job = tid + rep * NTHREADS;   // 0..2303 = 9 offsets x 256 channels
    int i = job >> 8;
    int c = job & 255;
    int row = h + i - DMAX;
    bool rowok = (row >= 0) && (row < H_);
    const float* src = FM1 + (((size_t)(b * C_ + c) * H_ + (rowok ? row : 0)) * W_);
#pragma unroll
    for (int g = 0; g < 6; ++g) {
      int wc = w0 - DMAX + g * 4;     // multiple of 4 -> aligned when in range
      float vv[4] = {0.f, 0.f, 0.f, 0.f};
      if (rowok) {
        if (wc >= 0 && wc <= W_ - 4) {
          float4 v = *(const float4*)(src + wc);
          vv[0] = v.x; vv[1] = v.y; vv[2] = v.z; vv[3] = v.w;
        } else {
#pragma unroll
          for (int e = 0; e < 4; ++e) {
            int w = wc + e;
            if (w >= 0 && w < W_) vv[e] = src[w];
          }
        }
      }
#pragma unroll
      for (int e = 0; e < 4; ++e) {
        int n = g * 4 + e;
        unsigned short hi, lo;
        split2(vv[e], hi, lo);
        size_t off = (size_t)(i * NCOL + n) * KP + c;
        sBhi[off] = hi;
        sBlo[off] = lo;
      }
    }
  }

  __syncthreads();

  // ---- Compute: wave wv owns vertical offset i = wv; two N-tiles of 16 cols.
  const int wv    = tid >> 5;          // 0..8
  const int lane  = tid & 31;
  const int lhalf = (lane < 16) ? 0 : 1;
  const int lrow  = lane & 15;
  const int i     = wv;

  v8f acc0 = {0.f,0.f,0.f,0.f,0.f,0.f,0.f,0.f};
  v8f acc1 = {0.f,0.f,0.f,0.f,0.f,0.f,0.f,0.f};

  const int col0    = lrow;                       // N-tile 0: cols 0..15
  const int col1raw = 16 + lrow;                  // N-tile 1: cols 16..31
  const int col1    = (col1raw < NCOL) ? col1raw : 0;  // clamp: garbage cols never stored

  const unsigned short* aBaseHi  = sAhi + lrow * KP;
  const unsigned short* aBaseLo  = sAlo + lrow * KP;
  const unsigned short* b0BaseHi = sBhi + (size_t)(i * NCOL + col0) * KP;
  const unsigned short* b0BaseLo = sBlo + (size_t)(i * NCOL + col0) * KP;
  const unsigned short* b1BaseHi = sBhi + (size_t)(i * NCOL + col1) * KP;
  const unsigned short* b1BaseLo = sBlo + (size_t)(i * NCOL + col1) * KP;

  // 16-bit k order used here (contiguous, same for A and B):
  //  A (16x32): lanes<16 -> K k0..k0+15 contiguous;     lanes>=16 -> k0+16..k0+31
  //  B (32x16): lanes<16 -> K k0..k0+15 contiguous;     lanes>=16 -> k0+16..k0+31
  const int aK = lhalf ? 16 : 0;
  const int bK = lhalf ? 16 : 0;

#pragma unroll
  for (int k0 = 0; k0 < C_; k0 += 32) {
    Frag ahi, alo, b0h, b0l, b1h, b1l;
    const unsigned short* p;

    p = aBaseHi + k0 + aK;  ahi.q[0] = *(const uint4*)(p); ahi.q[1] = *(const uint4*)(p + 8);
    p = aBaseLo + k0 + aK;  alo.q[0] = *(const uint4*)(p); alo.q[1] = *(const uint4*)(p + 8);

    p = b0BaseHi + k0 + bK; b0h.q[0] = *(const uint4*)(p); b0h.q[1] = *(const uint4*)(p + 8);
    p = b0BaseLo + k0 + bK; b0l.q[0] = *(const uint4*)(p); b0l.q[1] = *(const uint4*)(p + 8);
    p = b1BaseHi + k0 + bK; b1h.q[0] = *(const uint4*)(p); b1h.q[1] = *(const uint4*)(p + 8);
    p = b1BaseLo + k0 + bK; b1l.q[0] = *(const uint4*)(p); b1l.q[1] = *(const uint4*)(p + 8);

    // Split-bf16 fp32 emulation: D += Ahi*Bhi + Ahi*Blo + Alo*Bhi
    acc0 = __builtin_amdgcn_wmma_f32_16x16x32_bf16(false, ahi.v, false, b0h.v, (short)0, acc0, false, false);
    acc0 = __builtin_amdgcn_wmma_f32_16x16x32_bf16(false, ahi.v, false, b0l.v, (short)0, acc0, false, false);
    acc0 = __builtin_amdgcn_wmma_f32_16x16x32_bf16(false, alo.v, false, b0h.v, (short)0, acc0, false, false);

    acc1 = __builtin_amdgcn_wmma_f32_16x16x32_bf16(false, ahi.v, false, b1h.v, (short)0, acc1, false, false);
    acc1 = __builtin_amdgcn_wmma_f32_16x16x32_bf16(false, ahi.v, false, b1l.v, (short)0, acc1, false, false);
    acc1 = __builtin_amdgcn_wmma_f32_16x16x32_bf16(false, alo.v, false, b1h.v, (short)0, acc1, false, false);
  }

  // ---- Scatter band elements: D tile element (m, n) -> out[b,h,w0+m,i,j], j = col - m
  // C/D layout: lane<16 -> N=lane, M=r; lane>=16 -> N=lane-16, M=r+8
#pragma unroll
  for (int r = 0; r < 8; ++r) {
    int m = r + lhalf * 8;
    size_t base = ((((size_t)b * H_ + h) * W_ + (w0 + m)) * DD + i) * DD;
    int j0 = lrow - m;              // tile 0: col = lrow
    if (j0 >= 0 && j0 < DD) out[base + j0] = acc0[r];
    int j1 = 16 + lrow - m;         // tile 1: col = 16 + lrow
    if (j1 >= 0 && j1 < DD) out[base + j1] = acc1[r];
  }
}

extern "C" void kernel_launch(void* const* d_in, const int* in_sizes, int n_in,
                              void* d_out, int out_size, void* d_ws, size_t ws_size,
                              hipStream_t stream) {
  const float* FM0 = (const float*)d_in[0];
  const float* FM1 = (const float*)d_in[1];
  float* out = (float*)d_out;

  const size_t shmem_bytes =
      (size_t)(2 * WT * KP + 2 * DD * NCOL * KP) * sizeof(unsigned short);  // ~245 KB

  hipFuncSetAttribute((const void*)corr_wmma_kernel,
                      hipFuncAttributeMaxDynamicSharedMemorySize, (int)shmem_bytes);

  dim3 grid(W_ / WT, H_, B_);   // (6, 96, 8) = 4608 workgroups
  corr_wmma_kernel<<<grid, NTHREADS, shmem_bytes, stream>>>(FM0, FM1, out);
}
